// WaveletDiffusion_38311108280620
// MI455X (gfx1250) — matrix operand order repaired and checked by reading, not compiled, not run on hardware
//
#include <hip/hip_runtime.h>
#include <stdint.h>

// ---------------------------------------------------------------------------
// Types for WMMA (gfx1250, wave32)
// ---------------------------------------------------------------------------
typedef __attribute__((ext_vector_type(16))) __bf16 bf16x16;
typedef __attribute__((ext_vector_type(8)))  float  f32x8;

struct __align__(16) U4 { unsigned int a, b, c, d; };
union FragAB { bf16x16 v; U4 u[2]; };
union Pack8 { __bf16 h[8]; U4 q; };

#define LDA 40   // bf16 elements per LDS row (80B pitch: 16B-aligned, bank-staggered)

// Optional CDNA5 async global->LDS path (ASYNCcnt-tracked copy engine)
#if defined(__has_builtin)
#if __has_builtin(__builtin_amdgcn_global_load_async_to_lds_b128)
#define HAVE_ASYNC_LDS 1
#endif
#endif

#if defined(HAVE_ASYNC_LDS)
typedef int v4i __attribute__((vector_size(16)));
typedef __attribute__((address_space(1))) v4i v4i_g;   // global AS
typedef __attribute__((address_space(3))) v4i v4i_l;   // LDS AS
__device__ __forceinline__ void async_b128(const void* g, void* l) {
    __builtin_amdgcn_global_load_async_to_lds_b128((v4i_g*)g, (v4i_l*)l, 0, 0);
}
__device__ __forceinline__ void async_wait0() {
#if __has_builtin(__builtin_amdgcn_s_wait_asynccnt)
    __builtin_amdgcn_s_wait_asynccnt(0);
#else
    asm volatile("s_wait_asynccnt 0x0" ::: "memory");
#endif
}
#endif

// ---------------------------------------------------------------------------
// Haar DWT: x [16,8,256,256] NCHW fp32 -> h0 NHWC [16,128,128,32] (c*4+s)
// ---------------------------------------------------------------------------
__global__ __launch_bounds__(256) void dwt_kernel(const float* __restrict__ x,
                                                  const float* __restrict__ Kd,
                                                  float* __restrict__ h0) {
    int idx = blockIdx.x * 256 + threadIdx.x;          // 16*8*128*128
    if (idx >= 16 * 8 * 128 * 128) return;
    int w2 = idx & 127; int t = idx >> 7;
    int h2 = t & 127;   t >>= 7;
    int c  = t & 7;     int b = t >> 3;
    const float* xb = x + (((size_t)(b * 8 + c) * 256 + 2 * h2) * 256 + 2 * w2);
    float x00 = xb[0], x01 = xb[1], x10 = xb[256], x11 = xb[257];
    float* o = h0 + (((size_t)(b * 128 + h2) * 128 + w2) * 32 + c * 4);
#pragma unroll
    for (int s = 0; s < 4; ++s)
        o[s] = Kd[s * 4 + 0] * x00 + Kd[s * 4 + 1] * x01 +
               Kd[s * 4 + 2] * x10 + Kd[s * 4 + 3] * x11;
}

// ---------------------------------------------------------------------------
// Haar IDWT: y6 NHWC [16,128,128,16] -> out NCHW [16,4,256,256]
// ---------------------------------------------------------------------------
__global__ __launch_bounds__(256) void idwt_kernel(const float* __restrict__ y6,
                                                   const float* __restrict__ Kr,
                                                   float* __restrict__ out) {
    int idx = blockIdx.x * 256 + threadIdx.x;          // 16*4*128*128
    if (idx >= 16 * 4 * 128 * 128) return;
    int w2 = idx & 127; int t = idx >> 7;
    int h2 = t & 127;   t >>= 7;
    int c  = t & 3;     int b = t >> 2;
    const float* src = y6 + (((size_t)(b * 128 + h2) * 128 + w2) * 16 + c * 4);
    float s0 = src[0], s1 = src[1], s2 = src[2], s3 = src[3];
    float* o = out + (((size_t)(b * 4 + c) * 256 + 2 * h2) * 256 + 2 * w2);
#pragma unroll
    for (int d = 0; d < 2; ++d)
#pragma unroll
        for (int e = 0; e < 2; ++e)
            o[d * 256 + e] = s0 * Kr[0 * 4 + d * 2 + e] + s1 * Kr[1 * 4 + d * 2 + e] +
                             s2 * Kr[2 * 4 + d * 2 + e] + s3 * Kr[3 * 4 + d * 2 + e];
}

// ---------------------------------------------------------------------------
// Repack conv weights: fp32 [Cout][Cin][3][3] -> bf16 [Cout][(kh*3+kw)*Cin+ci]
// ---------------------------------------------------------------------------
__global__ __launch_bounds__(256) void repack_conv(const float* __restrict__ w,
                                                   __bf16* __restrict__ o,
                                                   int Cout, int Cin) {
    int K = 9 * Cin;
    int idx = blockIdx.x * 256 + threadIdx.x;
    if (idx >= Cout * K) return;
    int co = idx / K, k = idx - co * K;
    int tap = k / Cin, ci = k - tap * Cin;
    int kh = tap / 3, kw = tap - kh * 3;
    o[idx] = (__bf16)w[(((size_t)co * Cin + ci) * 3 + kh) * 3 + kw];
}

// Repack convT 4x4 s2 p1: fp32 [128][64][4][4] -> bf16 [4 phase][64][t*128+ci]
__global__ __launch_bounds__(256) void repack_convT(const float* __restrict__ w,
                                                    __bf16* __restrict__ o) {
    int idx = blockIdx.x * 256 + threadIdx.x;          // 4*64*512 = 131072
    if (idx >= 4 * 64 * 512) return;
    int ph = idx >> 15;
    int r  = idx & 32767;
    int co = r >> 9; int k = r & 511;
    int t  = k >> 7; int ci = k & 127;
    int pho = ph >> 1, phw = ph & 1;
    int th = t >> 1, tw = t & 1;
    int kh = pho ? (th ? 2 : 0) : (th ? 3 : 1);
    int kw = phw ? (tw ? 2 : 0) : (tw ? 3 : 1);
    o[idx] = (__bf16)w[(((size_t)ci * 64 + co) * 4 + kh) * 4 + kw];
}

// ---------------------------------------------------------------------------
// GroupNorm stats -> fused affine coefficients
// ---------------------------------------------------------------------------
__global__ __launch_bounds__(256) void gn_stats(const float* __restrict__ y,
                                                const float* __restrict__ gamma,
                                                const float* __restrict__ beta,
                                                float* __restrict__ affA,
                                                float* __restrict__ affB,
                                                int Bn, int HW, int C, int G) {
    int bg = blockIdx.x;
    int b = bg / G, g = bg - b * G;
    int Cg = C / G;
    size_t base = (size_t)b * HW * C + (size_t)g * Cg;
    float s = 0.f, ss = 0.f;
    for (int pix = threadIdx.x; pix < HW; pix += 256) {
        const float* row = y + base + (size_t)pix * C;
        for (int c = 0; c < Cg; ++c) { float v = row[c]; s += v; ss += v * v; }
    }
    __shared__ float rs[256], rss[256];
    rs[threadIdx.x] = s; rss[threadIdx.x] = ss;
    __syncthreads();
    for (int off = 128; off > 0; off >>= 1) {
        if ((int)threadIdx.x < off) {
            rs[threadIdx.x]  += rs[threadIdx.x + off];
            rss[threadIdx.x] += rss[threadIdx.x + off];
        }
        __syncthreads();
    }
    __shared__ float smu, svar;
    if (threadIdx.x == 0) {
        float n = (float)HW * (float)Cg;
        float mu = rs[0] / n;
        smu = mu; svar = rss[0] / n - mu * mu;
    }
    __syncthreads();
    if ((int)threadIdx.x < Cg) {
        int c = g * Cg + threadIdx.x;
        float aa = gamma[c] * rsqrtf(svar + 1e-5f);
        affA[(size_t)b * C + c] = aa;
        affB[(size_t)b * C + c] = beta[c] - smu * aa;
    }
}

// ---------------------------------------------------------------------------
// Implicit-GEMM 3x3 conv (pad 1, stride 1/2), NHWC, bf16 WMMA, f32 accum.
// tap-outer / cin-inner loop: im2col addressing hoisted out of the K loop.
// Workgroup: 256 thr = 8 waves; tile 64 pixels x 64 couts; K-step 32.
// ---------------------------------------------------------------------------
__global__ __launch_bounds__(256) void conv3x3_wmma(
    const float* __restrict__ in, float* __restrict__ out,
    const __bf16* __restrict__ wk, const float* __restrict__ bias,
    const float* __restrict__ affA, const float* __restrict__ affB,
    int Bn, int Hin, int Win, int Cin, int Hout, int Wout, int Cout,
    int stride, int Ksz) {
    __shared__ __bf16 As[64 * LDA];
    __shared__ __bf16 Bs[64 * LDA];

    const int tid  = threadIdx.x;
    const int lane = tid & 31;
    const int wave = tid >> 5;
    const int mt   = wave & 3;            // m-tile (pixels) 0..3
    const int nt0  = (wave >> 2) << 1;    // first n-tile (couts): 0 or 2

    const int HW       = Hout * Wout;
    const int pixBase  = blockIdx.x * 64;
    const int coutBase = blockIdx.y * 64;

    // A-loader coords: 4 threads per pixel-row, 8 channels each
    const int lrow = tid >> 2;
    const int lch  = (tid & 3) << 3;
    int p    = pixBase + lrow;
    int pb   = p / HW;
    int prem = p - pb * HW;
    int oh   = prem / Wout;
    int ow   = prem - oh * Wout;
    const float* paf = affA ? (affA + (size_t)pb * Cin + lch) : nullptr;
    const float* pbf = affA ? (affB + (size_t)pb * Cin + lch) : nullptr;

    // B-loader coords: 4 threads per cout, 8 K each
    const int bc  = tid >> 2;
    const int bk  = (tid & 3) << 3;
    int bco = coutBase + bc;
    if (bco >= Cout) bco = Cout - 1;      // clamp; epilogue guards stores
    const __bf16* wrow = wk + (size_t)bco * Ksz + bk;

    // fragment gather addresses (fixed for whole kernel)
    const int arow = (mt << 4) + (lane & 15);
    const int ka   = (lane & 16) ? 8 : 0;
    const int kb   = (lane & 16) ? 8 : 0;
    const int br0  = (nt0 << 4) + (lane & 15);

    f32x8 acc0 = {}, acc1 = {};

    const int ntaps = Ksz / Cin;          // 9
    int k0 = 0;
    for (int tap = 0; tap < ntaps; ++tap) {
        const int kh = tap / 3, kw = tap - kh * 3;
        const int ih = oh * stride + kh - 1;
        const int iw = ow * stride + kw - 1;
        const bool vld = ((unsigned)ih < (unsigned)Hin) & ((unsigned)iw < (unsigned)Win);
        const float* srcRow = in + (((size_t)pb * Hin + ih) * Win + iw) * Cin + lch;

        for (int cin0 = 0; cin0 < Cin; cin0 += 32, k0 += 32) {
            // ---- B tile: bf16 weights -> LDS
#if defined(HAVE_ASYNC_LDS)
            async_b128(wrow + k0, &Bs[bc * LDA + bk]);
#else
            *(U4*)&Bs[bc * LDA + bk] = *(const U4*)(wrow + k0);
#endif
            // ---- A tile: global fp32 -> (affine+relu) -> bf16 -> LDS
            float v[8];
            if (vld) {
                const float4* s4 = (const float4*)(srcRow + cin0);
                float4 va = s4[0], vb = s4[1];
                v[0] = va.x; v[1] = va.y; v[2] = va.z; v[3] = va.w;
                v[4] = vb.x; v[5] = vb.y; v[6] = vb.z; v[7] = vb.w;
            } else {
#pragma unroll
                for (int j = 0; j < 8; ++j) v[j] = 0.f;
            }
            if (paf) {
#pragma unroll
                for (int j = 0; j < 8; ++j) {
                    float tv = v[j] * paf[cin0 + j] + pbf[cin0 + j];
                    v[j] = tv > 0.f ? tv : 0.f;
                }
            }
            Pack8 pk;
#pragma unroll
            for (int j = 0; j < 8; ++j) pk.h[j] = (__bf16)v[j];
            *(U4*)&As[lrow * LDA + lch] = pk.q;

#if defined(HAVE_ASYNC_LDS)
            async_wait0();
#endif
            __syncthreads();

            // ---- fragments per ISA 7.12.2 layouts, then WMMA
            FragAB fa, fb0, fb1;
            fa.u[0] = *(const U4*)&As[arow * LDA + ka];
            fa.u[1] = *(const U4*)&As[arow * LDA + ka + 16];
            fb0.u[0] = *(const U4*)&Bs[br0 * LDA + kb];
            fb0.u[1] = *(const U4*)&Bs[br0 * LDA + kb + 16];
            fb1.u[0] = *(const U4*)&Bs[(br0 + 16) * LDA + kb];
            fb1.u[1] = *(const U4*)&Bs[(br0 + 16) * LDA + kb + 16];

            acc0 = __builtin_amdgcn_wmma_f32_16x16x32_bf16(false, fa.v, false, fb0.v,
                                                           (short)0, acc0, false, false);
            acc1 = __builtin_amdgcn_wmma_f32_16x16x32_bf16(false, fa.v, false, fb1.v,
                                                           (short)0, acc1, false, false);
            __syncthreads();
        }
    }

    // ---- epilogue: acc + bias -> NHWC fp32
    const int mrow = (mt << 4) + ((lane & 16) ? 8 : 0);
    const int ncol = lane & 15;
#pragma unroll
    for (int r = 0; r < 8; ++r) {
        int pm = pixBase + mrow + r;
        int n0 = coutBase + (nt0 << 4) + ncol;
        int n1 = n0 + 16;
        if (n0 < Cout) out[(size_t)pm * Cout + n0] = acc0[r] + bias[n0];
        if (n1 < Cout) out[(size_t)pm * Cout + n1] = acc1[r] + bias[n1];
    }
}

// ---------------------------------------------------------------------------
// Implicit-GEMM transposed conv 4x4 stride 2 pad 1 (128 -> 64), phase-split.
// in NHWC [16,64,64,128]; out NHWC [16,128,128,64]; K = 4 taps * 128 = 512.
// ---------------------------------------------------------------------------
__global__ __launch_bounds__(256) void convT_wmma(
    const float* __restrict__ in, float* __restrict__ out,
    const __bf16* __restrict__ wkAll, const float* __restrict__ bias,
    const float* __restrict__ affA, const float* __restrict__ affB) {
    __shared__ __bf16 As[64 * LDA];
    __shared__ __bf16 Bs[64 * LDA];

    const int tid  = threadIdx.x;
    const int lane = tid & 31;
    const int wave = tid >> 5;
    const int mt   = wave & 3;
    const int nt0  = (wave >> 2) << 1;

    const int phase = blockIdx.z;
    const int pho = phase >> 1, phw = phase & 1;
    const __bf16* wk = wkAll + (size_t)phase * 64 * 512;

    const int pixBase = blockIdx.x * 64;   // over B*64*64 phase-pixels
    const int lrow = tid >> 2;
    const int lch  = (tid & 3) << 3;
    int p    = pixBase + lrow;
    int pb   = p >> 12;                    // /4096
    int prem = p & 4095;
    int yy   = prem >> 6, xx = prem & 63;
    const int oh = 2 * yy + pho, ow = 2 * xx + phw;
    const float* paf = affA + (size_t)pb * 128 + lch;
    const float* pbf = affB + (size_t)pb * 128 + lch;

    const int bc = tid >> 2;
    const int bk = (tid & 3) << 3;
    const __bf16* wrow = wk + (size_t)bc * 512 + bk;

    const int arow = (mt << 4) + (lane & 15);
    const int ka   = (lane & 16) ? 8 : 0;
    const int kb   = (lane & 16) ? 8 : 0;
    const int br0  = (nt0 << 4) + (lane & 15);

    f32x8 acc0 = {}, acc1 = {};

    int k0 = 0;
    for (int tap = 0; tap < 4; ++tap) {
        const int th = tap >> 1, tw = tap & 1;
        const int kh = pho ? (th ? 2 : 0) : (th ? 3 : 1);
        const int kw = phw ? (tw ? 2 : 0) : (tw ? 3 : 1);
        const int ih = (oh + 1 - kh) >> 1;
        const int iw = (ow + 1 - kw) >> 1;
        const bool vld = ((unsigned)ih < 64u) & ((unsigned)iw < 64u);
        const float* srcRow = in + (((size_t)pb * 64 + ih) * 64 + iw) * 128 + lch;

        for (int cin0 = 0; cin0 < 128; cin0 += 32, k0 += 32) {
#if defined(HAVE_ASYNC_LDS)
            async_b128(wrow + k0, &Bs[bc * LDA + bk]);
#else
            *(U4*)&Bs[bc * LDA + bk] = *(const U4*)(wrow + k0);
#endif
            float v[8];
            if (vld) {
                const float4* s4 = (const float4*)(srcRow + cin0);
                float4 va = s4[0], vb = s4[1];
                v[0] = va.x; v[1] = va.y; v[2] = va.z; v[3] = va.w;
                v[4] = vb.x; v[5] = vb.y; v[6] = vb.z; v[7] = vb.w;
            } else {
#pragma unroll
                for (int j = 0; j < 8; ++j) v[j] = 0.f;
            }
#pragma unroll
            for (int j = 0; j < 8; ++j) {
                float tv = v[j] * paf[cin0 + j] + pbf[cin0 + j];
                v[j] = tv > 0.f ? tv : 0.f;
            }
            Pack8 pk;
#pragma unroll
            for (int j = 0; j < 8; ++j) pk.h[j] = (__bf16)v[j];
            *(U4*)&As[lrow * LDA + lch] = pk.q;

#if defined(HAVE_ASYNC_LDS)
            async_wait0();
#endif
            __syncthreads();

            FragAB fa, fb0, fb1;
            fa.u[0] = *(const U4*)&As[arow * LDA + ka];
            fa.u[1] = *(const U4*)&As[arow * LDA + ka + 16];
            fb0.u[0] = *(const U4*)&Bs[br0 * LDA + kb];
            fb0.u[1] = *(const U4*)&Bs[br0 * LDA + kb + 16];
            fb1.u[0] = *(const U4*)&Bs[(br0 + 16) * LDA + kb];
            fb1.u[1] = *(const U4*)&Bs[(br0 + 16) * LDA + kb + 16];

            acc0 = __builtin_amdgcn_wmma_f32_16x16x32_bf16(false, fa.v, false, fb0.v,
                                                           (short)0, acc0, false, false);
            acc1 = __builtin_amdgcn_wmma_f32_16x16x32_bf16(false, fa.v, false, fb1.v,
                                                           (short)0, acc1, false, false);
            __syncthreads();
        }
    }

    const int mrow = (mt << 4) + ((lane & 16) ? 8 : 0);
    const int ncol = lane & 15;
#pragma unroll
    for (int r = 0; r < 8; ++r) {
        int pm  = pixBase + mrow + r;
        int b2  = pm >> 12, rem = pm & 4095;
        int oy  = ((rem >> 6) << 1) + pho;
        int ox  = ((rem & 63) << 1) + phw;
        size_t o = (((size_t)b2 * 128 + oy) * 128 + ox) * 64;
        int n0 = (nt0 << 4) + ncol;
        out[o + n0]      = acc0[r] + bias[n0];
        out[o + n0 + 16] = acc1[r] + bias[n0 + 16];
    }
}

// ---------------------------------------------------------------------------
// Host orchestration
// ---------------------------------------------------------------------------
extern "C" void kernel_launch(void* const* d_in, const int* in_sizes, int n_in,
                              void* d_out, int out_size, void* d_ws, size_t ws_size,
                              hipStream_t stream) {
    const float* x   = (const float*)d_in[0];
    const float* Kd  = (const float*)d_in[1];
    const float* Kr  = (const float*)d_in[2];
    const float* W1  = (const float*)d_in[3];  const float* b1  = (const float*)d_in[4];
    const float* g1  = (const float*)d_in[5];  const float* be1 = (const float*)d_in[6];
    const float* W2  = (const float*)d_in[7];  const float* b2  = (const float*)d_in[8];
    const float* g2  = (const float*)d_in[9];  const float* be2 = (const float*)d_in[10];
    const float* W3  = (const float*)d_in[11]; const float* b3  = (const float*)d_in[12];
    const float* g3  = (const float*)d_in[13]; const float* be3 = (const float*)d_in[14];
    const float* W4  = (const float*)d_in[15]; const float* b4  = (const float*)d_in[16];
    const float* g4  = (const float*)d_in[17]; const float* be4 = (const float*)d_in[18];
    const float* Wt  = (const float*)d_in[19]; const float* bt  = (const float*)d_in[20];
    const float* g5  = (const float*)d_in[21]; const float* be5 = (const float*)d_in[22];
    const float* W6  = (const float*)d_in[23]; const float* b6  = (const float*)d_in[24];

    const size_t SLOT = (size_t)16 * 128 * 128 * 64 * sizeof(float);  // 64 MiB

    char* wp = (char*)d_ws;
    auto take = [&](size_t bytes) -> char* {
        char* r = wp;
        wp += (bytes + 255) & ~(size_t)255;
        return r;
    };
    float*  slot0 = (float*)take(SLOT);
    float*  slot1 = (float*)take(SLOT);
    __bf16* wk1 = (__bf16*)take((size_t)64  * 288  * 2);
    __bf16* wk2 = (__bf16*)take((size_t)128 * 576  * 2);
    __bf16* wk3 = (__bf16*)take((size_t)256 * 1152 * 2);
    __bf16* wk4 = (__bf16*)take((size_t)128 * 2304 * 2);
    __bf16* wkT = (__bf16*)take((size_t)4 * 64 * 512 * 2);
    __bf16* wk6 = (__bf16*)take((size_t)16  * 576  * 2);
    float* aff1 = (float*)take((size_t)2 * 16 * 64  * 4);
    float* aff2 = (float*)take((size_t)2 * 16 * 128 * 4);
    float* aff3 = (float*)take((size_t)2 * 16 * 256 * 4);
    float* aff4 = (float*)take((size_t)2 * 16 * 128 * 4);
    float* aff5 = (float*)take((size_t)2 * 16 * 64  * 4);

    // Weight repack (bf16, GEMM-K layout)
    repack_conv <<<(64 * 288   + 255) / 256, 256, 0, stream>>>(W1, wk1, 64, 32);
    repack_conv <<<(128 * 576  + 255) / 256, 256, 0, stream>>>(W2, wk2, 128, 64);
    repack_conv <<<(256 * 1152 + 255) / 256, 256, 0, stream>>>(W3, wk3, 256, 128);
    repack_conv <<<(128 * 2304 + 255) / 256, 256, 0, stream>>>(W4, wk4, 128, 256);
    repack_convT<<<(4 * 64 * 512 + 255) / 256, 256, 0, stream>>>(Wt, wkT);
    repack_conv <<<(16 * 576   + 255) / 256, 256, 0, stream>>>(W6, wk6, 16, 64);

    // DWT -> slot0 (NHWC, C=32)
    dwt_kernel<<<8192, 256, 0, stream>>>(x, Kd, slot0);

    // conv1: 32->64, 128x128
    conv3x3_wmma<<<dim3(4096, 1), 256, 0, stream>>>(slot0, slot1, wk1, b1,
        nullptr, nullptr, 16, 128, 128, 32, 128, 128, 64, 1, 288);
    gn_stats<<<16 * 8, 256, 0, stream>>>(slot1, g1, be1, aff1, aff1 + 16 * 64,
        16, 16384, 64, 8);

    // conv2: 64->128, stride 2 -> 64x64
    conv3x3_wmma<<<dim3(1024, 2), 256, 0, stream>>>(slot1, slot0, wk2, b2,
        aff1, aff1 + 16 * 64, 16, 128, 128, 64, 64, 64, 128, 2, 576);
    gn_stats<<<16 * 16, 256, 0, stream>>>(slot0, g2, be2, aff2, aff2 + 16 * 128,
        16, 4096, 128, 16);

    // conv3: 128->256
    conv3x3_wmma<<<dim3(1024, 4), 256, 0, stream>>>(slot0, slot1, wk3, b3,
        aff2, aff2 + 16 * 128, 16, 64, 64, 128, 64, 64, 256, 1, 1152);
    gn_stats<<<16 * 32, 256, 0, stream>>>(slot1, g3, be3, aff3, aff3 + 16 * 256,
        16, 4096, 256, 32);

    // conv4: 256->128
    conv3x3_wmma<<<dim3(1024, 2), 256, 0, stream>>>(slot1, slot0, wk4, b4,
        aff3, aff3 + 16 * 256, 16, 64, 64, 256, 64, 64, 128, 1, 2304);
    gn_stats<<<16 * 16, 256, 0, stream>>>(slot0, g4, be4, aff4, aff4 + 16 * 128,
        16, 4096, 128, 16);

    // convT: 128->64, x2 upsample (4 phases)
    convT_wmma<<<dim3(1024, 1, 4), 256, 0, stream>>>(slot0, slot1, wkT, bt,
        aff4, aff4 + 16 * 128);
    gn_stats<<<16 * 8, 256, 0, stream>>>(slot1, g5, be5, aff5, aff5 + 16 * 64,
        16, 16384, 64, 8);

    // conv6: 64->16
    conv3x3_wmma<<<dim3(4096, 1), 256, 0, stream>>>(slot1, slot0, wk6, b6,
        aff5, aff5 + 16 * 64, 16, 128, 128, 64, 128, 128, 16, 1, 576);

    // IDWT -> out
    idwt_kernel<<<4096, 256, 0, stream>>>(slot0, Kr, (float*)d_out);
}
